// CausalTGAT_1915555414536
// MI455X (gfx1250) — compile-verified, matrix operand order checked by reading, not run on hardware
//
#include <hip/hip_runtime.h>
#include <hip/hip_bf16.h>
#include <math.h>

typedef _Float16 h16;
typedef __attribute__((ext_vector_type(16))) _Float16 v16h;
typedef __attribute__((ext_vector_type(8)))  _Float16 v8h;
typedef __attribute__((ext_vector_type(8)))  float    v8f;

#define B_    128
#define T_    256
#define NF_   17
#define T2V_  16
#define GH_   32
#define NH_   4
#define GRUH_ 128
#define D0PAD 64
#define GO_   128                 // NH*GH
#define NROWS (B_*T_*NF_)         // 557056
#define NG    (B_*T_)             // 32768

union AFrag { v16h v; v8h h2[2]; };

// fast transcendental helpers (v_exp_f32 / v_rcp_f32 paths, TRANS32 pipe)
__device__ __forceinline__ float fexp(float x)   { return __expf(x); }
__device__ __forceinline__ float frcp(float x)   { return __builtin_amdgcn_rcpf(x); }
__device__ __forceinline__ float fsigmoid(float x){ return frcp(1.f + fexp(-x)); }
__device__ __forceinline__ float ftanh(float x)  { return 1.f - 2.f * frcp(1.f + fexp(2.f * x)); }

// ---------------------------------------------------------------------------
// K1: build node features nf = [x, mask, t2v(times), t2v(delta)] -> f16, K padded to 64
// ---------------------------------------------------------------------------
__global__ __launch_bounds__(256) void build_nf_kernel(
    const float* __restrict__ X, const float* __restrict__ mask,
    const float* __restrict__ delta, const float* __restrict__ times,
    const float* __restrict__ aw0, const float* __restrict__ ab0,
    const float* __restrict__ aW,  const float* __restrict__ aB,
    const float* __restrict__ dw0, const float* __restrict__ db0,
    const float* __restrict__ dW,  const float* __restrict__ dB,
    h16* __restrict__ A)
{
    int r = blockIdx.x * blockDim.x + threadIdx.x;
    if (r >= NROWS) return;
    int f = r % NF_;
    int g = r / NF_;                 // b*T + t
    float tv = times[g];
    float dv = delta[(size_t)g * NF_ + f];
    h16* row = A + (size_t)r * D0PAD;
    row[0] = (h16)X[(size_t)g * NF_ + f];
    row[1] = (h16)mask[(size_t)g * NF_ + f];
    row[2] = (h16)(tv * aw0[0] + ab0[0]);
    #pragma unroll
    for (int k = 0; k < T2V_ - 1; ++k) row[3 + k] = (h16)__sinf(tv * aW[k] + aB[k]);
    row[2 + T2V_] = (h16)(dv * dw0[0] + db0[0]);
    #pragma unroll
    for (int k = 0; k < T2V_ - 1; ++k) row[3 + T2V_ + k] = (h16)__sinf(dv * dW[k] + dB[k]);
    #pragma unroll
    for (int k = 2 + 2 * T2V_; k < D0PAD; ++k) row[k] = (h16)0.f;
}

// ---------------------------------------------------------------------------
// K2: parameter conversion kernels (f32 -> f16, optionally transposed/padded)
// dst layout: [N][Kpad] with dst[n][k] = src[k][n] (zero-padded past K)
// ---------------------------------------------------------------------------
__global__ __launch_bounds__(256) void conv_pad_transpose_kernel(
    const float* __restrict__ src, h16* __restrict__ dst, int K, int N, int Kpad)
{
    int idx = blockIdx.x * blockDim.x + threadIdx.x;
    if (idx >= N * Kpad) return;
    int n = idx / Kpad, k = idx % Kpad;
    dst[idx] = (k < K) ? (h16)src[(size_t)k * N + n] : (h16)0.f;
}

__global__ __launch_bounds__(256) void conv_f32_to_f16_kernel(
    const float* __restrict__ src, h16* __restrict__ dst, int count)
{
    int idx = blockIdx.x * blockDim.x + threadIdx.x;
    if (idx < count) dst[idx] = (h16)src[idx];
}

// ---------------------------------------------------------------------------
// K3: WMMA GEMM  C[M,N](f16) = A[M,K](f16) @ Wt^T  where Wt is [N][K] f16.
// K is a compile-time constant so all fragments stay in VGPRs (no scratch).
// Block: 256 thr (8 waves), 128 rows x 64-col chunk (grid.y = N/64).
// ---------------------------------------------------------------------------
template <int K>
__global__ __launch_bounds__(256) void gemm_wmma_kernel(
    const h16* __restrict__ A, const h16* __restrict__ Wt,
    h16* __restrict__ C, int N)
{
    constexpr int NKT = K / 32;
    __shared__ h16 wt_s[64 * K];       // <= 32 KB (K <= 256)
    int tid  = threadIdx.x;
    int wave = tid >> 5, lane = tid & 31;
    int colchunk = blockIdx.y * 64;

    for (int idx = tid; idx < 64 * K; idx += 256) {
        int n = idx / K, k = idx % K;
        wt_s[n * K + k] = Wt[(size_t)(colchunk + n) * K + k];
    }
    __syncthreads();

    int rowbase = blockIdx.x * 128 + wave * 16;
    int m  = lane & 15;
    int kh = lane >> 4;

    AFrag afr[NKT];
    #pragma unroll
    for (int kt = 0; kt < NKT; ++kt) {
        const h16* p = A + (size_t)(rowbase + m) * K + kt * 32 + kh * 8;
        afr[kt].h2[0] = *(const v8h*)p;
        afr[kt].h2[1] = *(const v8h*)(p + 16);
    }

    #pragma unroll
    for (int nt = 0; nt < 4; ++nt) {
        v8f acc = {};
        #pragma unroll
        for (int kt = 0; kt < NKT; ++kt) {
            AFrag bfr;
            const h16* p = wt_s + (nt * 16 + m) * K + kt * 32 + kh * 8;
            bfr.h2[0] = *(const v8h*)p;
            bfr.h2[1] = *(const v8h*)(p + 16);
            acc = __builtin_amdgcn_wmma_f32_16x16x32_f16(
                false, afr[kt].v, false, bfr.v, (short)0, acc, false, false);
        }
        int col = colchunk + nt * 16 + m;
        #pragma unroll
        for (int v = 0; v < 8; ++v) {
            int row = rowbase + v + 8 * kh;
            C[(size_t)row * N + col] = (h16)acc[v];
        }
    }
}

// ---------------------------------------------------------------------------
// K4: GAT attention (per 17-node graph). 128 thr = 4 waves, 1 graph/wave.
// layer==0: writes elu(attn@p) to Hout.  layer==1: residual + elu + mean -> Zout.
// ---------------------------------------------------------------------------
__global__ __launch_bounds__(128) void gat_attention_kernel(
    const h16* __restrict__ P, const float* __restrict__ adj,
    const float* __restrict__ a_src, const float* __restrict__ a_dst,
    const h16* __restrict__ Hin, h16* __restrict__ Hout, h16* __restrict__ Zout,
    int layer)
{
    __shared__ h16   p_s[4][NF_ * GO_];
    __shared__ float e_s[4][NH_ * NF_ * NF_];
    __shared__ float es_s[4][NH_ * NF_];
    __shared__ float ed_s[4][NH_ * NF_];

    int tid = threadIdx.x;
    int wave = tid >> 5, lane = tid & 31;
    int g = blockIdx.x * 4 + wave;
    const h16* pg = P + (size_t)g * NF_ * GO_;

    for (int idx = lane; idx < NF_ * GO_; idx += 32) p_s[wave][idx] = pg[idx];
    __syncthreads();

    for (int pr = lane; pr < NH_ * NF_; pr += 32) {
        int hh = pr / NF_, i = pr % NF_;
        float se = 0.f, sd = 0.f;
        #pragma unroll
        for (int o = 0; o < GH_; ++o) {
            float pv = (float)p_s[wave][i * GO_ + hh * GH_ + o];
            se += pv * a_src[hh * GH_ + o];
            sd += pv * a_dst[hh * GH_ + o];
        }
        es_s[wave][pr] = se;
        ed_s[wave][pr] = sd;
    }
    __syncthreads();

    for (int idx = lane; idx < NH_ * NF_ * NF_; idx += 32) {
        int hh  = idx / (NF_ * NF_);
        int rem = idx % (NF_ * NF_);
        int i = rem / NF_, j = rem % NF_;
        float e = es_s[wave][hh * NF_ + i] + ed_s[wave][hh * NF_ + j];
        e = (e > 0.f) ? e : 0.2f * e;
        if (adj[i * NF_ + j] <= 0.f) e = -1.0e9f;
        e_s[wave][idx] = e;
    }
    __syncthreads();

    for (int pr = lane; pr < NH_ * NF_; pr += 32) {
        float* rowp = &e_s[wave][pr * NF_];
        float mx = rowp[0];
        for (int j = 1; j < NF_; ++j) mx = fmaxf(mx, rowp[j]);
        float s = 0.f;
        for (int j = 0; j < NF_; ++j) { float t = fexp(rowp[j] - mx); rowp[j] = t; s += t; }
        float inv = frcp(s);
        for (int j = 0; j < NF_; ++j) rowp[j] *= inv;
    }
    __syncthreads();

    #pragma unroll
    for (int cc = 0; cc < 4; ++cc) {
        int c  = lane * 4 + cc;
        int hh = c / GH_;
        float zacc = 0.f;
        for (int i = 0; i < NF_; ++i) {
            const float* arow = &e_s[wave][(hh * NF_ + i) * NF_];
            float acc = 0.f;
            #pragma unroll
            for (int j = 0; j < NF_; ++j) acc += arow[j] * (float)p_s[wave][j * GO_ + c];
            if (layer) acc += (float)Hin[(size_t)(g * NF_ + i) * GO_ + c];
            acc = (acc > 0.f) ? acc : (fexp(acc) - 1.f);
            if (layer) zacc += acc;
            else       Hout[(size_t)(g * NF_ + i) * GO_ + c] = (h16)acc;
        }
        if (layer) Zout[(size_t)g * GO_ + c] = (h16)(zacc * (1.f / (float)NF_));
    }
}

// ---------------------------------------------------------------------------
// K5: persistent GRU scan, one block per direction. Hidden state lives in LDS
// (f16, 32 KB). Each wave owns a 16-row batch strip; per step it computes the
// recurrent GEMM gh = h @ whh^T with WMMA: for each 16-col tile it evaluates
// the r/z/n gate tiles (whh row groups 0/128/256 + jcol) so the three f32
// accumulators are lane-aligned and gate math stays in registers. Gate
// nonlinearities use the native v_exp_f32/v_rcp_f32 TRANS pipe so they
// co-execute with the WMMAs of the next tile.
// ---------------------------------------------------------------------------
__global__ __launch_bounds__(256) void gru_scan_kernel(
    const h16* __restrict__ gx, const h16* __restrict__ whh16,
    const float* __restrict__ bih, const float* __restrict__ bhh,
    h16* __restrict__ y16, float* __restrict__ yout, float* __restrict__ hT,
    int reverse, int layer, int dir)
{
    __shared__ h16 hbuf[B_ * GRUH_];   // 32 KB
    int tid = threadIdx.x;
    int wave = tid >> 5, lane = tid & 31;
    for (int idx = tid; idx < B_ * GRUH_; idx += 256) hbuf[idx] = (h16)0.f;
    __syncthreads();

    int rowbase = wave * 16;
    int m  = lane & 15;
    int kh = lane >> 4;

    for (int s = 0; s < T_; ++s) {
        int t = reverse ? (T_ - 1 - s) : s;

        AFrag afr[4];
        #pragma unroll
        for (int kt = 0; kt < 4; ++kt) {
            const h16* p = &hbuf[(rowbase + m) * GRUH_ + kt * 32 + kh * 8];
            afr[kt].h2[0] = *(const v8h*)p;
            afr[kt].h2[1] = *(const v8h*)(p + 16);
        }
        // prefetch next timestep's gx slab for this strip
        {
            int tn = reverse ? (t > 0 ? t - 1 : 0) : (t < T_ - 1 ? t + 1 : t);
            __builtin_prefetch(&gx[((size_t)rowbase * T_ + tn) * 384], 0, 1);
        }

        for (int nt = 0; nt < 8; ++nt) {
            v8f cr = {}, cz = {}, cn = {};
            int jcol = nt * 16 + m;
            #pragma unroll
            for (int kt = 0; kt < 4; ++kt) {
                AFrag br, bz, bn;
                const h16* p0 = whh16 + (size_t)(      jcol) * GRUH_ + kt * 32 + kh * 8;
                const h16* p1 = whh16 + (size_t)(128 + jcol) * GRUH_ + kt * 32 + kh * 8;
                const h16* p2 = whh16 + (size_t)(256 + jcol) * GRUH_ + kt * 32 + kh * 8;
                br.h2[0] = *(const v8h*)p0; br.h2[1] = *(const v8h*)(p0 + 16);
                bz.h2[0] = *(const v8h*)p1; bz.h2[1] = *(const v8h*)(p1 + 16);
                bn.h2[0] = *(const v8h*)p2; bn.h2[1] = *(const v8h*)(p2 + 16);
                cr = __builtin_amdgcn_wmma_f32_16x16x32_f16(false, afr[kt].v, false, br.v, (short)0, cr, false, false);
                cz = __builtin_amdgcn_wmma_f32_16x16x32_f16(false, afr[kt].v, false, bz.v, (short)0, cz, false, false);
                cn = __builtin_amdgcn_wmma_f32_16x16x32_f16(false, afr[kt].v, false, bn.v, (short)0, cn, false, false);
            }
            float bihr = bih[jcol], bihz = bih[128 + jcol], bihn = bih[256 + jcol];
            float bhhr = bhh[jcol], bhhz = bhh[128 + jcol], bhhn = bhh[256 + jcol];
            #pragma unroll
            for (int v = 0; v < 8; ++v) {
                int b = rowbase + v + 8 * kh;
                size_t gb = ((size_t)b * T_ + t) * 384;
                float gxr = (float)gx[gb + jcol]       + bihr;
                float gxz = (float)gx[gb + 128 + jcol] + bihz;
                float gxn = (float)gx[gb + 256 + jcol] + bihn;
                float r = fsigmoid(gxr + cr[v] + bhhr);
                float z = fsigmoid(gxz + cz[v] + bhhz);
                float n = ftanh(gxn + r * (cn[v] + bhhn));
                float hold = (float)hbuf[b * GRUH_ + jcol];
                float h2 = (1.f - z) * n + z * hold;
                hbuf[b * GRUH_ + jcol] = (h16)h2;
                size_t yb = ((size_t)b * T_ + t) * 256 + (size_t)dir * 128 + jcol;
                if (layer == 0) y16[yb]  = (h16)h2;
                else            yout[yb] = h2;
            }
        }
        __syncthreads();
    }

    for (int idx = tid; idx < B_ * GRUH_; idx += 256) hT[idx] = (float)hbuf[idx];
}

// ---------------------------------------------------------------------------
// K6: output heads. 1 block of 128 threads, one batch row per thread.
// ---------------------------------------------------------------------------
__global__ __launch_bounds__(128) void heads_kernel(
    const float* __restrict__ hTf, const float* __restrict__ hTb,
    const float* __restrict__ mh_w1, const float* __restrict__ mh_b1,
    const float* __restrict__ mh_w2, const float* __restrict__ mh_b2,
    const float* __restrict__ ch_w1, const float* __restrict__ ch_b1,
    const float* __restrict__ ch_w2, const float* __restrict__ ch_b2,
    float* __restrict__ out)
{
    int b = threadIdx.x;
    float* out_m    = out;
    float* out_sig  = out + 128;
    float* out_risk = out + 256;
    float* out_pe   = out + 384;

    for (int k = 0; k < 128; ++k) {
        out_pe[b * 256 + k]       = hTf[b * 128 + k];
        out_pe[b * 256 + 128 + k] = hTb[b * 128 + k];
    }
    float mval = mh_b2[0], rv = ch_b2[0];
    for (int u = 0; u < 64; ++u) {
        float s1 = mh_b1[u], s2 = ch_b1[u];
        for (int k = 0; k < 128; ++k) {
            float pf = hTf[b * 128 + k];
            float pb = hTb[b * 128 + k];
            s1 += pf * mh_w1[u * 256 + k] + pb * mh_w1[u * 256 + 128 + k];
            s2 += pf * ch_w1[u * 256 + k] + pb * ch_w1[u * 256 + 128 + k];
        }
        mval += fmaxf(s1, 0.f) * mh_w2[u];
        rv   += fmaxf(s2, 0.f) * ch_w2[u];
    }
    out_m[b]    = mval;
    out_sig[b]  = fsigmoid(mval);
    out_risk[b] = rv;
}

// ---------------------------------------------------------------------------
// host launch
// ---------------------------------------------------------------------------
extern "C" void kernel_launch(void* const* d_in, const int* in_sizes, int n_in,
                              void* d_out, int out_size, void* d_ws, size_t ws_size,
                              hipStream_t stream)
{
    (void)in_sizes; (void)n_in; (void)out_size; (void)ws_size;
    const float* X     = (const float*)d_in[0];
    const float* mask  = (const float*)d_in[1];
    const float* delta = (const float*)d_in[2];
    const float* times = (const float*)d_in[3];
    const float* adj   = (const float*)d_in[4];
    const float* aw0 = (const float*)d_in[5],  *ab0 = (const float*)d_in[6];
    const float* aW  = (const float*)d_in[7],  *aB  = (const float*)d_in[8];
    const float* dw0 = (const float*)d_in[9],  *db0 = (const float*)d_in[10];
    const float* dW  = (const float*)d_in[11], *dB  = (const float*)d_in[12];
    const float* gW0  = (const float*)d_in[13];
    const float* gas0 = (const float*)d_in[14], *gad0 = (const float*)d_in[15];
    const float* gW1  = (const float*)d_in[16];
    const float* gas1 = (const float*)d_in[17], *gad1 = (const float*)d_in[18];
    const float* g0f_wih = (const float*)d_in[19], *g0f_whh = (const float*)d_in[20];
    const float* g0f_bih = (const float*)d_in[21], *g0f_bhh = (const float*)d_in[22];
    const float* g0b_wih = (const float*)d_in[23], *g0b_whh = (const float*)d_in[24];
    const float* g0b_bih = (const float*)d_in[25], *g0b_bhh = (const float*)d_in[26];
    const float* g1f_wih = (const float*)d_in[27], *g1f_whh = (const float*)d_in[28];
    const float* g1f_bih = (const float*)d_in[29], *g1f_bhh = (const float*)d_in[30];
    const float* g1b_wih = (const float*)d_in[31], *g1b_whh = (const float*)d_in[32];
    const float* g1b_bih = (const float*)d_in[33], *g1b_bhh = (const float*)d_in[34];
    const float* mh_w1 = (const float*)d_in[35], *mh_b1 = (const float*)d_in[36];
    const float* mh_w2 = (const float*)d_in[37], *mh_b2 = (const float*)d_in[38];
    const float* ch_w1 = (const float*)d_in[39], *ch_b1 = (const float*)d_in[40];
    const float* ch_w2 = (const float*)d_in[41], *ch_b2 = (const float*)d_in[42];

    // workspace carve-out (all 256B aligned)
    char* ws = (char*)d_ws;
    size_t off = 0;
    auto carve = [&](size_t bytes) { char* p = ws + off; off = (off + bytes + 255) & ~(size_t)255; return p; };
    h16* A_nf   = (h16*)carve((size_t)NROWS * D0PAD * 2);     // 71.3 MB
    h16* P      = (h16*)carve((size_t)NROWS * GO_ * 2);       // 142.6 MB (p0, then p1)
    h16* H1     = (h16*)carve((size_t)NROWS * GO_ * 2);       // 142.6 MB
    h16* Z      = (h16*)carve((size_t)NG * GO_ * 2);          // 8.4 MB
    h16* GXa    = (h16*)carve((size_t)NG * 384 * 2);          // 25.2 MB (reused layer1)
    h16* GXb    = (h16*)carve((size_t)NG * 384 * 2);          // 25.2 MB
    h16* Y16    = (h16*)carve((size_t)NG * 256 * 2);          // 16.8 MB
    h16* W0t    = (h16*)carve((size_t)128 * 64 * 2);
    h16* W1t    = (h16*)carve((size_t)128 * 128 * 2);
    h16* wih0f  = (h16*)carve((size_t)384 * 128 * 2);
    h16* wih0b  = (h16*)carve((size_t)384 * 128 * 2);
    h16* wih1f  = (h16*)carve((size_t)384 * 256 * 2);
    h16* wih1b  = (h16*)carve((size_t)384 * 256 * 2);
    h16* whh0f  = (h16*)carve((size_t)384 * 128 * 2);
    h16* whh0b  = (h16*)carve((size_t)384 * 128 * 2);
    h16* whh1f  = (h16*)carve((size_t)384 * 128 * 2);
    h16* whh1b  = (h16*)carve((size_t)384 * 128 * 2);
    float* hT0f = (float*)carve((size_t)B_ * GRUH_ * 4);
    float* hT0b = (float*)carve((size_t)B_ * GRUH_ * 4);
    float* hT1f = (float*)carve((size_t)B_ * GRUH_ * 4);
    float* hT1b = (float*)carve((size_t)B_ * GRUH_ * 4);

    float* out_f   = (float*)d_out;
    float* out_gru = out_f + 384 + (size_t)B_ * 256;   // m(128)+sig(128)+risk(128)+pe(32768)

    // 1. node features
    build_nf_kernel<<<(NROWS + 255) / 256, 256, 0, stream>>>(
        X, mask, delta, times, aw0, ab0, aW, aB, dw0, db0, dW, dB, A_nf);

    // 2. parameter conversion
    conv_pad_transpose_kernel<<<(128 * 64 + 255) / 256, 256, 0, stream>>>(gW0, W0t, 34, 128, 64);
    conv_pad_transpose_kernel<<<(128 * 128 + 255) / 256, 256, 0, stream>>>(gW1, W1t, 128, 128, 128);
    conv_f32_to_f16_kernel<<<(384 * 128 + 255) / 256, 256, 0, stream>>>(g0f_wih, wih0f, 384 * 128);
    conv_f32_to_f16_kernel<<<(384 * 128 + 255) / 256, 256, 0, stream>>>(g0b_wih, wih0b, 384 * 128);
    conv_f32_to_f16_kernel<<<(384 * 256 + 255) / 256, 256, 0, stream>>>(g1f_wih, wih1f, 384 * 256);
    conv_f32_to_f16_kernel<<<(384 * 256 + 255) / 256, 256, 0, stream>>>(g1b_wih, wih1b, 384 * 256);
    conv_f32_to_f16_kernel<<<(384 * 128 + 255) / 256, 256, 0, stream>>>(g0f_whh, whh0f, 384 * 128);
    conv_f32_to_f16_kernel<<<(384 * 128 + 255) / 256, 256, 0, stream>>>(g0b_whh, whh0b, 384 * 128);
    conv_f32_to_f16_kernel<<<(384 * 128 + 255) / 256, 256, 0, stream>>>(g1f_whh, whh1f, 384 * 128);
    conv_f32_to_f16_kernel<<<(384 * 128 + 255) / 256, 256, 0, stream>>>(g1b_whh, whh1b, 384 * 128);

    // 3. GAT layer 0: projection (K=64 padded) + attention
    gemm_wmma_kernel<64><<<dim3(NROWS / 128, 128 / 64), 256, 0, stream>>>(A_nf, W0t, P, 128);
    gat_attention_kernel<<<NG / 4, 128, 0, stream>>>(P, adj, gas0, gad0, H1, H1, Z, 0);

    // 4. GAT layer 1: projection (K=128) + attention (residual) + mean-pool
    gemm_wmma_kernel<128><<<dim3(NROWS / 128, 128 / 64), 256, 0, stream>>>(H1, W1t, P, 128);
    gat_attention_kernel<<<NG / 4, 128, 0, stream>>>(P, adj, gas1, gad1, H1, H1, Z, 1);

    // 5. GRU layer 0: input GEMMs then scans (fwd/bwd)
    gemm_wmma_kernel<128><<<dim3(NG / 128, 384 / 64), 256, 0, stream>>>(Z, wih0f, GXa, 384);
    gemm_wmma_kernel<128><<<dim3(NG / 128, 384 / 64), 256, 0, stream>>>(Z, wih0b, GXb, 384);
    gru_scan_kernel<<<1, 256, 0, stream>>>(GXa, whh0f, g0f_bih, g0f_bhh, Y16, out_gru, hT0f, 0, 0, 0);
    gru_scan_kernel<<<1, 256, 0, stream>>>(GXb, whh0b, g0b_bih, g0b_bhh, Y16, out_gru, hT0b, 1, 0, 1);

    // 6. GRU layer 1: input GEMMs (K=256) then scans writing gru_out directly
    gemm_wmma_kernel<256><<<dim3(NG / 128, 384 / 64), 256, 0, stream>>>(Y16, wih1f, GXa, 384);
    gemm_wmma_kernel<256><<<dim3(NG / 128, 384 / 64), 256, 0, stream>>>(Y16, wih1b, GXb, 384);
    gru_scan_kernel<<<1, 256, 0, stream>>>(GXa, whh1f, g1f_bih, g1f_bhh, Y16, out_gru, hT1f, 0, 1, 0);
    gru_scan_kernel<<<1, 256, 0, stream>>>(GXb, whh1b, g1b_bih, g1b_bhh, Y16, out_gru, hT1b, 1, 1, 1);

    // 7. heads
    heads_kernel<<<1, 128, 0, stream>>>(hT1f, hT1b, mh_w1, mh_b1, mh_w2, mh_b2,
                                        ch_w1, ch_b1, ch_w2, ch_b2, out_f);
}
